// GraphAttentionLayer_76124000354826
// MI455X (gfx1250) — matrix operand order repaired and checked by reading, not compiled, not run on hardware
//
#include <hip/hip_runtime.h>
#include <hip/hip_bf16.h>

#define N_NODES 8192
#define D_IN    512
#define D_OUT   256
#define ALPHA   0.2f
#define MAXN    2048

typedef __attribute__((ext_vector_type(16))) __bf16 v16bf;
typedef __attribute__((ext_vector_type(8)))  float  v8f;

__device__ __forceinline__ __bf16 f2bf(float f) { return (__bf16)f; }

// -------------------------------------------------------------------------
// Kernel 1: h = features @ weight   (bf16 WMMA, f32 accumulate)
// One wave per 16x16 output tile, K-loop over 512 in steps of 32.
// -------------------------------------------------------------------------
__global__ void __launch_bounds__(128) gat_gemm_h(
    const float* __restrict__ X, const float* __restrict__ W,
    float* __restrict__ H)
{
  const int lane = threadIdx.x & 31;
  const int wave = threadIdx.x >> 5;
  const int tile = blockIdx.x * 4 + wave;
  const int NT   = D_OUT / 16;            // 16 tiles along N
  const int mt   = tile / NT;
  const int nt   = tile % NT;
  const int half = lane >> 4;
  const int mn   = lane & 15;
  const int row  = mt * 16 + mn;          // A row held by this lane
  const int col  = nt * 16 + mn;          // B/C column held by this lane

  v8f acc = {};
  for (int kb = 0; kb < D_IN; kb += 32) {
    // A fragment: element j -> K = kb + 16*(j/8) + 8*half + (j%8)
    const float* ap = X + (size_t)row * D_IN + kb + 8 * half;
    v16bf af;
#pragma unroll
    for (int r = 0; r < 8; ++r) af[r]     = f2bf(ap[r]);
#pragma unroll
    for (int r = 0; r < 8; ++r) af[8 + r] = f2bf(ap[16 + r]);
    // B fragment: element j -> K = kb + 16*(j/8) + 8*half + (j%8)  (row-major weight)
    const float* bp = W + (size_t)(kb + 8 * half) * D_OUT + col;
    v16bf bf;
#pragma unroll
    for (int j = 0; j < 16; ++j) bf[j] = f2bf(bp[(size_t)(j + (j & 8)) * D_OUT]);

    acc = __builtin_amdgcn_wmma_f32_16x16x32_bf16(false, af, false, bf,
                                                  (short)0, acc, false, false);
  }
  // C/D: element r -> (row mt*16 + 8*half + r, col)
  float* hp = H + (size_t)(mt * 16 + 8 * half) * D_OUT + col;
#pragma unroll
  for (int r = 0; r < 8; ++r) hp[(size_t)r * D_OUT] = acc[r];
}

// -------------------------------------------------------------------------
// Kernel 2: s_src = h @ a[:256], s_dst = h @ a[256:]   (one wave per row)
// -------------------------------------------------------------------------
__global__ void __launch_bounds__(256) gat_scores(
    const float* __restrict__ H, const float* __restrict__ av,
    float* __restrict__ s_src, float* __restrict__ s_dst)
{
  const int lane = threadIdx.x & 31;
  const int wave = threadIdx.x >> 5;
  const int row  = blockIdx.x * 8 + wave;
  const float* hr = H + (size_t)row * D_OUT;
  float ss = 0.f, sd = 0.f;
#pragma unroll
  for (int r = 0; r < 8; ++r) {
    int c = lane * 8 + r;
    float v = hr[c];
    ss += v * av[c];
    sd += v * av[D_OUT + c];
  }
#pragma unroll
  for (int off = 16; off > 0; off >>= 1) {
    ss += __shfl_xor(ss, off, 32);
    sd += __shfl_xor(sd, off, 32);
  }
  if (lane == 0) { s_src[row] = ss; s_dst[row] = sd; }
}

// -------------------------------------------------------------------------
// Kernel 3: fused masked-softmax + sparse aggregation + elu, one row/block.
// Adjacency row staged Global->LDS with CDNA5 async copies (ASYNCcnt),
// then deterministic neighbor extraction via count + LDS prefix sum.
// -------------------------------------------------------------------------
__global__ void __launch_bounds__(256) gat_row(
    const float* __restrict__ adj, const float* __restrict__ H,
    const float* __restrict__ s_src, const float* __restrict__ s_dst,
    float* __restrict__ out)
{
  __shared__ float sh_adj[N_NODES];     // 32 KB staged adjacency row
  __shared__ int   nbr[MAXN];
  __shared__ float wts[MAXN];
  __shared__ int   cnts[256];
  __shared__ float rbuf[256];

  const int i   = blockIdx.x;
  const int tid = threadIdx.x;
  const float* arow = adj + (size_t)i * N_NODES;
  const int base = tid * (N_NODES / 256);        // 32 columns per thread

  // ---- async stage: 8 x B128 per thread, global -> LDS, no VGPR round-trip
  // generic LDS pointer's low 32 bits == logical LDS byte address
#pragma unroll
  for (int r = 0; r < 32; r += 4) {
    unsigned lds_a = (unsigned)(uintptr_t)(const void*)(sh_adj + base + r);
    const float* ga = arow + base + r;
    asm volatile("global_load_async_to_lds_b128 %0, %1, off"
                 :: "v"(lds_a), "v"(ga) : "memory");
  }
  asm volatile("s_wait_asynccnt 0" ::: "memory");
  __syncthreads();

  // pass 1: count neighbors in this thread's chunk (from LDS)
  int c = 0;
#pragma unroll
  for (int r = 0; r < 32; r += 4) {
    float4 v = *reinterpret_cast<const float4*>(sh_adj + base + r);
    c += (v.x > 0.f) + (v.y > 0.f) + (v.z > 0.f) + (v.w > 0.f);
  }
  cnts[tid] = c;
  __syncthreads();
  // inclusive prefix sum (Hillis-Steele, 8 steps)
  for (int off = 1; off < 256; off <<= 1) {
    int v = (tid >= off) ? cnts[tid - off] : 0;
    __syncthreads();
    cnts[tid] += v;
    __syncthreads();
  }
  const int total = cnts[255];
  const int n = total < MAXN ? total : MAXN;
  int wp = cnts[tid] - c;
  // pass 2: emit indices in sorted (deterministic) order (from LDS)
  for (int r = 0; r < 32; ++r) {
    if (sh_adj[base + r] > 0.f) {
      if (wp < MAXN) nbr[wp] = base + r;
      ++wp;
    }
  }
  __syncthreads();

  // scores: leaky_relu(s_src[i] + s_dst[j])
  const float si = s_src[i];
  for (int t = tid; t < n; t += 256) {
    float x = si + s_dst[nbr[t]];
    wts[t] = (x >= 0.f) ? x : ALPHA * x;
  }
  __syncthreads();

  // row max
  float lm = -3.0e38f;
  for (int t = tid; t < n; t += 256) lm = fmaxf(lm, wts[t]);
  rbuf[tid] = lm;
  __syncthreads();
  for (int s = 128; s > 0; s >>= 1) {
    if (tid < s) rbuf[tid] = fmaxf(rbuf[tid], rbuf[tid + s]);
    __syncthreads();
  }
  const float m = rbuf[0];
  __syncthreads();

  // exp and sum
  float ls = 0.f;
  for (int t = tid; t < n; t += 256) {
    float e = __expf(wts[t] - m);
    wts[t] = e;
    ls += e;
  }
  __syncthreads();
  rbuf[tid] = ls;
  __syncthreads();
  for (int s = 128; s > 0; s >>= 1) {
    if (tid < s) rbuf[tid] += rbuf[tid + s];
    __syncthreads();
  }
  const float inv = 1.f / rbuf[0];

  // sparse aggregation: column `tid` of h' for row i, then elu
  float acc = 0.f;
  for (int k = 0; k < n; ++k)
    acc += wts[k] * H[(size_t)nbr[k] * D_OUT + tid];
  acc *= inv;
  out[(size_t)i * D_OUT + tid] = (acc > 0.f) ? acc : (__expf(acc) - 1.f);
}

// -------------------------------------------------------------------------
extern "C" void kernel_launch(void* const* d_in, const int* in_sizes, int n_in,
                              void* d_out, int out_size, void* d_ws, size_t ws_size,
                              hipStream_t stream) {
  (void)in_sizes; (void)n_in; (void)out_size; (void)ws_size;
  const float* adj    = (const float*)d_in[0];
  const float* feat   = (const float*)d_in[1];
  const float* weight = (const float*)d_in[2];
  const float* av     = (const float*)d_in[3];
  float* out = (float*)d_out;

  float* H     = (float*)d_ws;                       // 8192*256 f32 (8 MB)
  float* s_src = H + (size_t)N_NODES * D_OUT;        // 8192 f32
  float* s_dst = s_src + N_NODES;                    // 8192 f32

  const int tiles = (N_NODES / 16) * (D_OUT / 16);   // 8192 wave-tiles
  gat_gemm_h<<<tiles / 4, 128, 0, stream>>>(feat, weight, H);
  gat_scores<<<N_NODES / 8, 256, 0, stream>>>(H, av, s_src, s_dst);
  gat_row<<<N_NODES, 256, 0, stream>>>(adj, H, s_src, s_dst, out);
}
